// SelfAttention_42984032698623
// MI455X (gfx1250) — matrix operand order repaired and checked by reading, not compiled, not run on hardware
//
#include <hip/hip_runtime.h>

// ---------------------------------------------------------------------------
// Self-attention for MI455X (gfx1250, wave32, WMMA f32_16x16x32_f16)
//   B=4, N=4096, D=256.  H = softmax(QK^T/sqrt(D)) V,  Q/K/V = x @ W^T
//   Kernel 1: QKV projection GEMMs (V stored transposed for PV B-fragments).
//   Kernel 2: flash attention; 8 waves/WG cooperate on one batch; K/V tiles
//             double-buffered in LDS via GLOBAL_LOAD_ASYNC_TO_LDS_B128
//             (ASYNCcnt + workgroup barriers); P transposed through LDS.
// ---------------------------------------------------------------------------

typedef __attribute__((ext_vector_type(16))) _Float16       v16h;
typedef __attribute__((ext_vector_type(8)))  float          v8f;
typedef __attribute__((ext_vector_type(4)))  float          v4f;
typedef __attribute__((ext_vector_type(8)))  unsigned short v8us;

#define DDIM 256
#define NSEQ 4096
#define NBAT 4

union FragH {
    v16h h;
    v8us u[2];
};

#if __has_builtin(__builtin_amdgcn_s_wait_asynccnt)
#define WAIT_ASYNC16() __builtin_amdgcn_s_wait_asynccnt(16)
#else
#define WAIT_ASYNC16() asm volatile("s_wait_asynccnt 16" ::: "memory")
#endif

// One b128 per lane; INST_OFFSET applies to BOTH lds and global addresses
// (CDNA5 ISA), so 8 immediate offsets cover 128B/thread.
__device__ __forceinline__ void async_tile_b128x8(unsigned lds, unsigned long long gaddr) {
#pragma unroll
    for (int i = 0; i < 8; ++i)
        asm volatile("global_load_async_to_lds_b128 %0, %1, off offset:%2"
                     :: "v"(lds), "v"(gaddr), "n"(i * 16) : "memory");
}

// ---------------------------------------------------------------------------
// Kernel 1: Q/K/V projection.  y[n,e] = sum_d x[n,d] * W[e,d]
// grid = (256, 16, 3), block = 128 (4 waves); each wave -> one 16x16 tile.
// ---------------------------------------------------------------------------
__global__ void __launch_bounds__(128)
qkv_kernel(const float* __restrict__ x,
           const float* __restrict__ Wq,
           const float* __restrict__ Wk,
           const float* __restrict__ Wv,
           _Float16* __restrict__ Qh,
           _Float16* __restrict__ Kh,
           _Float16* __restrict__ Vt) {
    const int lane = threadIdx.x & 31;
    const int wave = threadIdx.x >> 5;
    const int r    = lane & 15;
    const int hl   = lane >> 4;
    const int m0   = (blockIdx.x * 4 + wave) * 16;
    const int n0   = blockIdx.y * 16;
    const int which = blockIdx.z;
    const float* W = (which == 0) ? Wq : (which == 1) ? Wk : Wv;

    v8f acc = {};
#pragma unroll
    for (int kc = 0; kc < 8; ++kc) {
        const int k0 = kc * 32;
        const float* ap = x + (size_t)(m0 + r) * DDIM + k0 + hl * 8;
        const float* bp = W + (size_t)(n0 + r) * DDIM + k0 + hl * 8;
        v4f a0 = *(const v4f*)(ap +  0);
        v4f a1 = *(const v4f*)(ap +  4);
        v4f a2 = *(const v4f*)(ap + 16);
        v4f a3 = *(const v4f*)(ap + 20);
        v4f b0 = *(const v4f*)(bp +  0);
        v4f b1 = *(const v4f*)(bp +  4);
        v4f b2 = *(const v4f*)(bp + 16);
        v4f b3 = *(const v4f*)(bp + 20);
        FragH a, b;
#pragma unroll
        for (int i = 0; i < 4; ++i) {
            a.h[i]      = (_Float16)a0[i];
            a.h[4 + i]  = (_Float16)a1[i];
            a.h[8 + i]  = (_Float16)a2[i];
            a.h[12 + i] = (_Float16)a3[i];
            b.h[i]      = (_Float16)b0[i];
            b.h[4 + i]  = (_Float16)b1[i];
            b.h[8 + i]  = (_Float16)b2[i];
            b.h[12 + i] = (_Float16)b3[i];
        }
        acc = __builtin_amdgcn_wmma_f32_16x16x32_f16(
            false, a.h, false, b.h, (short)0, acc, false, false);
    }

    if (which == 2) {
        const int rowbase = m0 + hl * 8;
        const int bb = rowbase >> 12;
        const int nn = rowbase & (NSEQ - 1);
        _Float16* vp = Vt + ((size_t)bb * DDIM + n0 + r) * NSEQ + nn;
#pragma unroll
        for (int v = 0; v < 8; ++v) vp[v] = (_Float16)acc[v];
    } else {
        _Float16* O = (which == 0) ? Qh : Kh;
#pragma unroll
        for (int v = 0; v < 8; ++v)
            O[(size_t)(m0 + v + 8 * hl) * DDIM + n0 + r] = (_Float16)acc[v];
    }
}

// ---------------------------------------------------------------------------
// Kernel 2: flash attention.  grid = 128, block = 256 (8 waves, one batch).
// Dynamic LDS: K tiles 2x32KB | V tiles 2x32KB | P staging 8x2KB = 144 KB.
// ---------------------------------------------------------------------------
__global__ void __launch_bounds__(256)
attn_kernel(const _Float16* __restrict__ Qh,
            const _Float16* __restrict__ Kh,
            const _Float16* __restrict__ Vt,
            float* __restrict__ H) {
    extern __shared__ _Float16 smem[];
    _Float16* kbuf = smem;                         // [2][64*256]
    _Float16* vbuf = smem + 2 * 64 * DDIM;         // [2][256*64]
    unsigned short* pball = (unsigned short*)(vbuf + 2 * DDIM * 64);

    const int t    = threadIdx.x;
    const int lane = t & 31;
    const int wave = t >> 5;
    const int r    = lane & 15;
    const int hl   = lane >> 4;
    const int b    = blockIdx.x >> 5;                       // 32 blocks / batch
    const int m0   = ((blockIdx.x & 31) * 8 + wave) * 16;   // query strip
    unsigned short* pbuf = pball + wave * 1024;

    // staging slices owned by this thread
    const int kkey = t >> 2;                 // 0..63
    const int kseg = (t & 3) * 64;           // 64 f16 = 128B
    const _Float16* kgp = Kh + ((size_t)(b * NSEQ + kkey)) * DDIM + kseg;
    const _Float16* vgp = Vt + ((size_t)(b * DDIM + t)) * NSEQ;
    const unsigned klds0 = (unsigned)(size_t)(kbuf + kkey * DDIM + kseg);
    const unsigned vlds0 = (unsigned)(size_t)(vbuf + t * 64);

    // resident Q fragments: 16 rows x 256 d
    FragH qf[8];
#pragma unroll
    for (int kc = 0; kc < 8; ++kc) {
        const _Float16* qp =
            Qh + ((size_t)(b * NSEQ + m0 + r)) * DDIM + kc * 32 + hl * 8;
        qf[kc].u[0] = *(const v8us*)qp;
        qf[kc].u[1] = *(const v8us*)(qp + 16);
    }

    v8f zero = {};
    v8f o[16];
#pragma unroll
    for (int i = 0; i < 16; ++i) o[i] = zero;
    float mi[8], li[8];
#pragma unroll
    for (int v = 0; v < 8; ++v) { mi[v] = -3.0e38f; li[v] = 0.0f; }

    // prologue: tile 0 -> buffer 0   (16 async instructions per wave)
    async_tile_b128x8(klds0, (unsigned long long)(size_t)kgp);
    async_tile_b128x8(vlds0, (unsigned long long)(size_t)vgp);

    for (int jb = 0; jb < NSEQ / 64; ++jb) {
        const int cur = jb & 1;
        // issue next tile into the other buffer (wraps on last iteration)
        {
            const int jn = ((jb + 1) & 63) * 64;
            const unsigned koff = (unsigned)((cur ^ 1) * 64 * DDIM * 2);
            async_tile_b128x8(klds0 + koff,
                (unsigned long long)(size_t)(kgp + (size_t)jn * DDIM));
            async_tile_b128x8(vlds0 + koff,
                (unsigned long long)(size_t)(vgp + jn));
        }
        WAIT_ASYNC16();      // oldest 16 (current tile) complete, in order
        __syncthreads();     // every wave's slice of the tile is in LDS

        const _Float16* ktile = kbuf + cur * 64 * DDIM;
        const _Float16* vtile = vbuf + cur * DDIM * 64;

        // ---- S = (Q K^T)/16 : 4 tiles of 16 keys, 8 WMMAs each ----
        v8f s[4];
#pragma unroll
        for (int tt = 0; tt < 4; ++tt) {
            v8f c = zero;
            const _Float16* kpb = ktile + (tt * 16 + r) * DDIM + hl * 8;
#pragma unroll
            for (int kc = 0; kc < 8; ++kc) {
                FragH kb;
                const _Float16* kp = kpb + kc * 32;
                kb.u[0] = *(const v8us*)kp;
                kb.u[1] = *(const v8us*)(kp + 16);
                c = __builtin_amdgcn_wmma_f32_16x16x32_f16(
                    false, qf[kc].h, false, kb.h, (short)0, c, false, false);
            }
            s[tt] = c * 0.0625f;   // 1/sqrt(256)
        }

        // ---- online softmax (rows live across 16 lanes of a half) ----
        float mnew[8], al[8];
#pragma unroll
        for (int v = 0; v < 8; ++v) {
            float tm = fmaxf(fmaxf(s[0][v], s[1][v]), fmaxf(s[2][v], s[3][v]));
            tm = fmaxf(tm, __shfl_xor(tm, 1));
            tm = fmaxf(tm, __shfl_xor(tm, 2));
            tm = fmaxf(tm, __shfl_xor(tm, 4));
            tm = fmaxf(tm, __shfl_xor(tm, 8));
            mnew[v] = fmaxf(mi[v], tm);
            al[v]   = __expf(mi[v] - mnew[v]);
            mi[v]   = mnew[v];
            li[v]  *= al[v];
        }
#pragma unroll
        for (int df = 0; df < 16; ++df)
#pragma unroll
            for (int v = 0; v < 8; ++v) o[df][v] *= al[v];

        // ---- P = exp(S - m); row sums; stage P (16x64 f16) in LDS ----
#pragma unroll
        for (int v = 0; v < 8; ++v) {
            float rs = 0.0f;
            const int row = v + hl * 8;
#pragma unroll
            for (int tt = 0; tt < 4; ++tt) {
                float p = __expf(s[tt][v] - mnew[v]);
                rs += p;
                _Float16 ph = (_Float16)p;
                unsigned short pu;
                __builtin_memcpy(&pu, &ph, 2);
                pbuf[row * 64 + tt * 16 + r] = pu;
            }
            rs += __shfl_xor(rs, 1);
            rs += __shfl_xor(rs, 2);
            rs += __shfl_xor(rs, 4);
            rs += __shfl_xor(rs, 8);
            li[v] += rs;
        }

        // ---- reload P as two 16x32 A-fragments ----
        FragH pa[2];
#pragma unroll
        for (int pc = 0; pc < 2; ++pc) {
            const unsigned short* pp = pbuf + r * 64 + pc * 32 + hl * 8;
            pa[pc].u[0] = *(const v8us*)pp;
            pa[pc].u[1] = *(const v8us*)(pp + 16);
        }

        // ---- O += P * V (V tile is d-major: contiguous key runs) ----
#pragma unroll
        for (int df = 0; df < 16; ++df) {
#pragma unroll
            for (int pc = 0; pc < 2; ++pc) {
                FragH vb;
                const _Float16* vp =
                    vtile + (df * 16 + r) * 64 + pc * 32 + hl * 8;
                vb.u[0] = *(const v8us*)vp;
                vb.u[1] = *(const v8us*)(vp + 16);
                o[df] = __builtin_amdgcn_wmma_f32_16x16x32_f16(
                    false, pa[pc].h, false, vb.h, (short)0, o[df], false, false);
            }
        }

        __syncthreads();     // all reads done before next tile overwrites
    }

    // ---- epilogue: O / l ----
    float inv[8];
#pragma unroll
    for (int v = 0; v < 8; ++v) inv[v] = 1.0f / li[v];
#pragma unroll
    for (int df = 0; df < 16; ++df)
#pragma unroll
        for (int v = 0; v < 8; ++v) {
            const int row = m0 + v + 8 * hl;
            H[((size_t)(b * NSEQ + row)) * DDIM + df * 16 + r] =
                o[df][v] * inv[v];
        }
}

// ---------------------------------------------------------------------------
extern "C" void kernel_launch(void* const* d_in, const int* in_sizes, int n_in,
                              void* d_out, int out_size, void* d_ws, size_t ws_size,
                              hipStream_t stream) {
    (void)in_sizes; (void)n_in; (void)out_size; (void)ws_size;
    const float* x  = (const float*)d_in[0];
    const float* Wq = (const float*)d_in[1];
    const float* Wk = (const float*)d_in[2];
    const float* Wv = (const float*)d_in[3];
    float* H = (float*)d_out;

    const size_t elems = (size_t)NBAT * NSEQ * DDIM;   // 4,194,304
    _Float16* Qh = (_Float16*)d_ws;
    _Float16* Kh = Qh + elems;
    _Float16* Vt = Kh + elems;                          // 25.2 MB total ws

    dim3 gq(256, 16, 3), bq(128);
    qkv_kernel<<<gq, bq, 0, stream>>>(x, Wq, Wk, Wv, Qh, Kh, Vt);

    // LDS: 2*32KB (K) + 2*32KB (V) + 16KB (P staging) = 147456 B
    const size_t lds_bytes = (size_t)(2 * 64 * DDIM + 2 * DDIM * 64) * sizeof(_Float16)
                           + 8 * 1024 * sizeof(unsigned short);
    dim3 ga(128), ba(256);
    attn_kernel<<<ga, ba, lds_bytes, stream>>>(Qh, Kh, Vt, H);
}
